// TransitionModel_42846593744943
// MI455X (gfx1250) — matrix operand order repaired and checked by reading, not compiled, not run on hardware
//
#include <hip/hip_runtime.h>
#include <hip/hip_bf16.h>
#include <math.h>

typedef __attribute__((ext_vector_type(8)))  _Float16 v8h;
typedef __attribute__((ext_vector_type(16))) _Float16 v16h;
typedef __attribute__((ext_vector_type(8)))  float    v8f;

#define N_DIM 1024
#define BATCH 16384

// ---------------------------------------------------------------------------
// Prologue A: column-wise logsumexp of W (softmax over axis=0).
// One thread per column; consecutive threads -> consecutive columns ->
// perfectly coalesced row-major sweeps of W (2 passes over 4 MB, L2-hot).
// ---------------------------------------------------------------------------
__global__ __launch_bounds__(256) void col_lse_kernel(
    const float* __restrict__ W, float* __restrict__ colLSE) {
  int k = blockIdx.x * 256 + threadIdx.x;  // 0..1023
  float mx = -INFINITY;
  for (int i = 0; i < N_DIM; ++i) mx = fmaxf(mx, W[(size_t)i * N_DIM + k]);
  float s = 0.f;
  for (int i = 0; i < N_DIM; ++i) s += __expf(W[(size_t)i * N_DIM + k] - mx);
  colLSE[k] = __logf(s) + mx;
}

// ---------------------------------------------------------------------------
// Prologue B: expA[i,k] = f16(exp(W[i,k] - colLSE[k] - mA[i])),
//             mA[i] = max_k (W[i,k] - colLSE[k]).  One block per row i.
// ---------------------------------------------------------------------------
__global__ __launch_bounds__(256) void expA_kernel(
    const float* __restrict__ W, const float* __restrict__ colLSE,
    _Float16* __restrict__ expA, float* __restrict__ mA) {
  int i = blockIdx.x;
  int t = threadIdx.x;
  float v[4];
  float mx = -INFINITY;
#pragma unroll
  for (int c = 0; c < 4; ++c) {
    int k = t + c * 256;
    v[c] = W[(size_t)i * N_DIM + k] - colLSE[k];
    mx = fmaxf(mx, v[c]);
  }
  __shared__ float red[256];
  red[t] = mx;
  __syncthreads();
  for (int s = 128; s > 0; s >>= 1) {
    if (t < s) red[t] = fmaxf(red[t], red[t + s]);
    __syncthreads();
  }
  float rm = red[0];
#pragma unroll
  for (int c = 0; c < 4; ++c)
    expA[(size_t)i * N_DIM + t + c * 256] = (_Float16)__expf(v[c] - rm);
  if (t == 0) mA[i] = rm;
}

// ---------------------------------------------------------------------------
// Prologue C: expB[b,k] = f16(exp(log_alpha[b,k] - mB[b])),
//             mB[b] = max_k log_alpha[b,k].  One block per batch row b.
// ---------------------------------------------------------------------------
__global__ __launch_bounds__(256) void expB_kernel(
    const float* __restrict__ la, _Float16* __restrict__ expB,
    float* __restrict__ mB) {
  int b = blockIdx.x;
  int t = threadIdx.x;
  float v[4];
  float mx = -INFINITY;
#pragma unroll
  for (int c = 0; c < 4; ++c) {
    v[c] = la[(size_t)b * N_DIM + t + c * 256];
    mx = fmaxf(mx, v[c]);
  }
  __shared__ float red[256];
  red[t] = mx;
  __syncthreads();
  for (int s = 128; s > 0; s >>= 1) {
    if (t < s) red[t] = fmaxf(red[t], red[t + s]);
    __syncthreads();
  }
  float rm = red[0];
#pragma unroll
  for (int c = 0; c < 4; ++c)
    expB[(size_t)b * N_DIM + t + c * 256] = (_Float16)__expf(v[c] - rm);
  if (t == 0) mB[b] = rm;
}

// ---------------------------------------------------------------------------
// Main GEMM + log epilogue.
//   out[m,n] = log( sum_k expB[m,k] * expA[n,k] ) + mB[m] + mA[n]
// Wave tile 64x64 (4x4 of 16x16), f16 WMMA with f32 accumulation.
// 16 WMMAs per 16 b128 loads per K-step (1.0 wmma/load, ~32 FLOP/B fetched).
// Block: 8 waves as 2(M) x 4(N) -> 128x256 tile. Grid (128, 4).
//
// Fragment layouts as used here (A/B contiguous k order is this code's choice) (lane = 32-wide, g = lane>>4, r = lane&15):
//   A (16x32, M x K): lane row M=r; halves 0..15 = K[g*16 .. g*16+15]
//                     (one contiguous 32B load)
//   B (32x16, K x N): lane col N=r; halves 0..15 = K[g*16 .. g*16+15]
//                     (one contiguous 32B load)
//   C/D (16x16 f32):  lane n = r; reg rr -> m = rr + 8*g
// ---------------------------------------------------------------------------
#define TM 4
#define TN 4

__global__ __launch_bounds__(256) void logmm_wmma_kernel(
    const _Float16* __restrict__ expB, const _Float16* __restrict__ expA,
    const float* __restrict__ mB, const float* __restrict__ mA,
    float* __restrict__ out) {
  const int K = N_DIM;
  int lane = threadIdx.x & 31;
  int wave = threadIdx.x >> 5;
  int wm = wave >> 2;  // 0..1
  int wn = wave & 3;   // 0..3
  int m0 = blockIdx.x * 128 + wm * 64;
  int n0 = blockIdx.y * 256 + wn * 64;
  int g = lane >> 4;
  int r = lane & 15;

  const _Float16* aRow[TM];
#pragma unroll
  for (int i = 0; i < TM; ++i) aRow[i] = expB + (size_t)(m0 + i * 16 + r) * K;
  const _Float16* bRow[TN];
#pragma unroll
  for (int j = 0; j < TN; ++j) bRow[j] = expA + (size_t)(n0 + j * 16 + r) * K;

  v8f acc[TM][TN] = {};

  for (int kb = 0; kb < K; kb += 32) {
    v16h af[TM];
    v16h bf[TN];
#pragma unroll
    for (int i = 0; i < TM; ++i) {
      af[i] = *(const v16h*)(aRow[i] + kb + g * 16);
    }
#pragma unroll
    for (int j = 0; j < TN; ++j)
      bf[j] = *(const v16h*)(bRow[j] + kb + g * 16);

#pragma unroll
    for (int i = 0; i < TM; ++i)
#pragma unroll
      for (int j = 0; j < TN; ++j)
        acc[i][j] = __builtin_amdgcn_wmma_f32_16x16x32_f16(
            false, af[i], false, bf[j], (short)0, acc[i][j], false, false);
  }

  // Epilogue: log(s) + mB[m] + mA[n]
#pragma unroll
  for (int i = 0; i < TM; ++i) {
#pragma unroll
    for (int j = 0; j < TN; ++j) {
      int n = n0 + j * 16 + r;
      float man = mA[n];
#pragma unroll
      for (int rr = 0; rr < 8; ++rr) {
        int m = m0 + i * 16 + rr + 8 * g;
        out[(size_t)m * N_DIM + n] = __logf(acc[i][j][rr]) + mB[m] + man;
      }
    }
  }
}

// ---------------------------------------------------------------------------
// Launch
// ---------------------------------------------------------------------------
extern "C" void kernel_launch(void* const* d_in, const int* in_sizes, int n_in,
                              void* d_out, int out_size, void* d_ws,
                              size_t ws_size, hipStream_t stream) {
  const float* log_alpha = (const float*)d_in[0];  // [BATCH, N]
  const float* W = (const float*)d_in[1];          // [N, N]
  float* out = (float*)d_out;                      // [BATCH, N]

  // Workspace layout (bytes):
  //   expB f16 [BATCH*N]  : 32 MB
  //   expA f16 [N*N]      :  2 MB
  //   mB   f32 [BATCH]    : 64 KB
  //   mA   f32 [N]        :  4 KB
  //   colLSE f32 [N]      :  4 KB
  char* ws = (char*)d_ws;
  _Float16* expB = (_Float16*)ws;
  _Float16* expA = (_Float16*)(ws + (size_t)BATCH * N_DIM * 2);
  float* mB = (float*)(ws + (size_t)BATCH * N_DIM * 2 + (size_t)N_DIM * N_DIM * 2);
  float* mA = (float*)((char*)mB + (size_t)BATCH * 4);
  float* colLSE = (float*)((char*)mA + (size_t)N_DIM * 4);

  col_lse_kernel<<<N_DIM / 256, 256, 0, stream>>>(W, colLSE);
  expA_kernel<<<N_DIM, 256, 0, stream>>>(W, colLSE, expA, mA);
  expB_kernel<<<BATCH, 256, 0, stream>>>(log_alpha, expB, mB);

  dim3 grid(BATCH / 128, N_DIM / 256);
  logmm_wmma_kernel<<<grid, 256, 0, stream>>>(expB, expA, mB, mA, out);
}
